// ParserTreeEncDP_76811195122065
// MI455X (gfx1250) — matrix operand order repaired and checked by reading, not compiled, not run on hardware
//
#include <hip/hip_runtime.h>
#include <hip/hip_bf16.h>

typedef __attribute__((ext_vector_type(16))) __bf16 v16bf;
typedef __attribute__((ext_vector_type(8)))  float  v8f;

#define B_     32
#define L_     512
#define H_     768
#define I_     3072
#define N_     1023            // 2*L-1
#define TOTAL_ 22
#define TO_    20
#define NT_    21
#define K_     4
#define M_     (B_ * N_)       // 32736 rows = 2046 x 16 = 682 x 48
#define EPS_   1e-7f

#define KT_    (H_ / 32)       // 24 K-tiles
#define NT16_  (I_ / 16)       // 192 16-col tiles of W1
#define MSUB_  3               // 16-row subtiles per block (48 rows)

// ---------------------------------------------------------------------------
// helpers
// ---------------------------------------------------------------------------
__device__ __forceinline__ float gelu_tanh(float x) {
    // jax.nn.gelu default (approximate=True)
    float x3 = x * x * x;
    float u  = 0.7978845608028654f * (x + 0.044715f * x3);
    return 0.5f * x * (1.0f + tanhf(u));
}

__device__ __forceinline__ void split_bf16(float f, __bf16& hi, __bf16& lo) {
    hi = (__bf16)f;
    lo = (__bf16)(f - (float)hi);
}

// A-fragment K index for element e, lane-half lh (contiguous k = 16*lh + e, same order as the B packing)
__device__ __forceinline__ int a_frag_k(int e, int lh) {
    return e + lh * 16;
}

#define WMMA_BF16(A, Bv, C) \
    __builtin_amdgcn_wmma_f32_16x16x32_bf16(false, (A), false, (Bv), (short)0, (C), false, false)

// ---------------------------------------------------------------------------
// Prep kernels: fp32 -> (hi,lo) bf16 split, pre-swizzled to WMMA fragment
// order so the GEMM does pure 128-bit packed loads.
// ---------------------------------------------------------------------------
__global__ __launch_bounds__(256) void prep_e_kernel(
    const float* __restrict__ E, __bf16* __restrict__ Ef_hi, __bf16* __restrict__ Ef_lo)
{
    size_t idx = (size_t)blockIdx.x * 256 + threadIdx.x;   // over M_*H_ elements
    if (idx >= (size_t)M_ * H_) return;
    int    e    = (int)(idx & 15);
    int    lane = (int)((idx >> 4) & 31);
    size_t tile = idx >> 9;
    int    kt   = (int)(tile % KT_);
    int    mt   = (int)(tile / KT_);
    int    lh   = (lane >= 16) ? 1 : 0;
    int    kk   = a_frag_k(e, lh);
    int    row  = mt * 16 + (lane & 15);
    float  f    = E[(size_t)row * H_ + kt * 32 + kk];
    __bf16 h, l; split_bf16(f, h, l);
    Ef_hi[idx] = h; Ef_lo[idx] = l;
}

__global__ __launch_bounds__(256) void prep_w_kernel(
    const float* __restrict__ W1, __bf16* __restrict__ Wf_hi, __bf16* __restrict__ Wf_lo)
{
    size_t idx = (size_t)blockIdx.x * 256 + threadIdx.x;   // over H_*I_ elements
    if (idx >= (size_t)H_ * I_) return;
    int    e    = (int)(idx & 15);
    int    lane = (int)((idx >> 4) & 31);
    size_t tile = idx >> 9;
    int    it16 = (int)(tile % NT16_);
    int    kt   = (int)(tile / NT16_);
    // B 32x16 in contiguous k order (same as A): lanes 0-15 hold K 0-15, lanes 16-31 hold K 16-31; N = lane&15
    int    kr   = kt * 32 + ((lane >= 16) ? 16 : 0) + e;
    int    nc   = it16 * 16 + (lane & 15);
    float  f    = W1[(size_t)kr * I_ + nc];
    __bf16 h, l; split_bf16(f, h, l);
    Wf_hi[idx] = h; Wf_lo[idx] = l;
}

// ---------------------------------------------------------------------------
// Fast fused MLP: 48-row M-tile per block (3 subtiles/wave) for 3x B-fragment
// reuse. Per k-step/wave: 10 packed b128-pair loads feed 18 v_wmma.
// ---------------------------------------------------------------------------
__global__ __launch_bounds__(256) void fused_mlp_pre_kernel(
    const __bf16* __restrict__ Ef_hi, const __bf16* __restrict__ Ef_lo,
    const __bf16* __restrict__ Wf_hi, const __bf16* __restrict__ Wf_lo,
    const float* __restrict__ b1, const float* __restrict__ W2,
    const float* __restrict__ b2, float* __restrict__ logits)
{
    __shared__ float hbuf[8][16][32];           // one 16-row subtile at a time
    __shared__ float lacc[8][16 * MSUB_][TOTAL_];

    const int tid  = threadIdx.x;
    const int w    = tid >> 5;
    const int lane = tid & 31;
    const int lh   = (lane >> 4) & 1;
    const int ml   = lane & 15;
    const int m0   = blockIdx.x * (16 * MSUB_);

    for (int t = lane; t < 16 * MSUB_ * TOTAL_; t += 32)
        lacc[w][t / TOTAL_][t % TOTAL_] = 0.0f;

    for (int itile = w; itile < I_ / 32; itile += 8) {
        const int icol = itile * 32;
        v8f c0[MSUB_], c1[MSUB_];
        #pragma unroll
        for (int st = 0; st < MSUB_; ++st) { c0[st] = (v8f){}; c1[st] = (v8f){}; }

        for (int kt = 0; kt < KT_; ++kt) {
            const size_t boff0 = ((size_t)(kt * NT16_ + itile * 2)     * 32 + lane) * 16;
            const size_t boff1 = ((size_t)(kt * NT16_ + itile * 2 + 1) * 32 + lane) * 16;
            v16bf b0h = *(const v16bf*)(Wf_hi + boff0);
            v16bf b0l = *(const v16bf*)(Wf_lo + boff0);
            v16bf b1h = *(const v16bf*)(Wf_hi + boff1);
            v16bf b1l = *(const v16bf*)(Wf_lo + boff1);

            #pragma unroll
            for (int st = 0; st < MSUB_; ++st) {
                const int mt = blockIdx.x * MSUB_ + st;   // global 16-row tile id
                const size_t aoff = ((size_t)(mt * KT_ + kt) * 32 + lane) * 16;
                v16bf a_hi = *(const v16bf*)(Ef_hi + aoff);
                v16bf a_lo = *(const v16bf*)(Ef_lo + aoff);

                c0[st] = WMMA_BF16(a_hi, b0h, c0[st]);
                c0[st] = WMMA_BF16(a_hi, b0l, c0[st]);
                c0[st] = WMMA_BF16(a_lo, b0h, c0[st]);
                c1[st] = WMMA_BF16(a_hi, b1h, c1[st]);
                c1[st] = WMMA_BF16(a_hi, b1l, c1[st]);
                c1[st] = WMMA_BF16(a_lo, b1h, c1[st]);
            }
        }

        const float bia0 = b1[icol + ml];
        const float bia1 = b1[icol + 16 + ml];

        #pragma unroll
        for (int st = 0; st < MSUB_; ++st) {
            #pragma unroll
            for (int r = 0; r < 8; ++r) {
                int row = r + lh * 8;
                hbuf[w][row][ml]      = gelu_tanh(c0[st][r] + bia0);
                hbuf[w][row][16 + ml] = gelu_tanh(c1[st][r] + bia1);
            }
            __syncthreads();   // uniform across all waves

            for (int t = lane; t < 16 * TOTAL_; t += 32) {
                int row = t / TOTAL_, cls = t % TOTAL_;
                float s = 0.0f;
                #pragma unroll
                for (int i = 0; i < 32; ++i)
                    s += hbuf[w][row][i] * W2[(size_t)(icol + i) * TOTAL_ + cls];
                lacc[w][st * 16 + row][cls] += s;
            }
            __syncthreads();
        }
    }

    __syncthreads();
    for (int t = tid; t < 16 * MSUB_ * TOTAL_; t += 256) {
        int row = t / TOTAL_, cls = t % TOTAL_;
        float s = b2[cls];
        #pragma unroll
        for (int ww = 0; ww < 8; ++ww) s += lacc[ww][row][cls];
        logits[(size_t)(m0 + row) * TOTAL_ + cls] = s;
    }
}

// ---------------------------------------------------------------------------
// Fallback fused MLP (inline conversions) — used only if ws is too small.
// ---------------------------------------------------------------------------
__global__ __launch_bounds__(256) void fused_mlp_kernel(
    const float* __restrict__ E,  const float* __restrict__ W1,
    const float* __restrict__ b1, const float* __restrict__ W2,
    const float* __restrict__ b2, float* __restrict__ logits)
{
    __shared__ float As[16][32];
    __shared__ float hbuf[8][16][32];
    __shared__ float lacc[8][16][TOTAL_];

    const int tid  = threadIdx.x;
    const int w    = tid >> 5;
    const int lane = tid & 31;
    const int lh   = (lane >> 4) & 1;
    const int ml   = lane & 15;
    const int m0   = blockIdx.x * 16;

    for (int t = lane; t < 16 * TOTAL_; t += 32)
        lacc[w][t / TOTAL_][t % TOTAL_] = 0.0f;

    for (int itile = w; itile < I_ / 32; itile += 8) {
        const int icol = itile * 32;
        v8f c0 = {}; v8f c1 = {};

        for (int k0 = 0; k0 < H_; k0 += 32) {
            __syncthreads();
            for (int t = tid; t < 512; t += 256) {
                int r = t >> 5, c = t & 31;
                As[r][c] = E[(size_t)(m0 + r) * H_ + k0 + c];
            }
            __syncthreads();

            v16bf a_hi, a_lo;
            #pragma unroll
            for (int e = 0; e < 16; ++e) {
                __bf16 h, l;
                split_bf16(As[ml][a_frag_k(e, lh)], h, l);
                a_hi[e] = h; a_lo[e] = l;
            }
            v16bf b0_hi, b0_lo, b1_hi, b1_lo;
            #pragma unroll
            for (int e = 0; e < 16; ++e) {
                int kk = e + lh * 16;
                const float* wrow = W1 + (size_t)(k0 + kk) * I_ + icol;
                __bf16 h, l;
                split_bf16(wrow[ml], h, l);      b0_hi[e] = h; b0_lo[e] = l;
                split_bf16(wrow[16 + ml], h, l); b1_hi[e] = h; b1_lo[e] = l;
            }
            c0 = WMMA_BF16(a_hi, b0_hi, c0);
            c0 = WMMA_BF16(a_hi, b0_lo, c0);
            c0 = WMMA_BF16(a_lo, b0_hi, c0);
            c1 = WMMA_BF16(a_hi, b1_hi, c1);
            c1 = WMMA_BF16(a_hi, b1_lo, c1);
            c1 = WMMA_BF16(a_lo, b1_hi, c1);
        }

        const float bia0 = b1[icol + ml];
        const float bia1 = b1[icol + 16 + ml];
        #pragma unroll
        for (int r = 0; r < 8; ++r) {
            int row = r + lh * 8;
            hbuf[w][row][ml]      = gelu_tanh(c0[r] + bia0);
            hbuf[w][row][16 + ml] = gelu_tanh(c1[r] + bia1);
        }
        __syncthreads();

        for (int t = lane; t < 16 * TOTAL_; t += 32) {
            int row = t / TOTAL_, cls = t % TOTAL_;
            float s = 0.0f;
            #pragma unroll
            for (int i = 0; i < 32; ++i)
                s += hbuf[w][row][i] * W2[(size_t)(icol + i) * TOTAL_ + cls];
            lacc[w][row][cls] += s;
        }
    }

    __syncthreads();
    for (int t = tid; t < 16 * TOTAL_; t += 256) {
        int row = t / TOTAL_, cls = t % TOTAL_;
        float s = b2[cls];
        #pragma unroll
        for (int ww = 0; ww < 8; ++ww) s += lacc[ww][row][cls];
        logits[(size_t)(blockIdx.x * 16 + row) * TOTAL_ + cls] = s;
    }
}

// ---------------------------------------------------------------------------
// Kernel 2: softmax + tree DP + loss. One block per batch element.
// ---------------------------------------------------------------------------
__device__ __forceinline__ void node_probs(const float* __restrict__ row,
                                           int t0, int t1, int t2, int t3,
                                           float tp[K_], float& nt, float& to)
{
    float mx = row[0];
    #pragma unroll
    for (int t = 1; t < TOTAL_; ++t) mx = fmaxf(mx, row[t]);
    float s = 0.0f;
    #pragma unroll
    for (int t = 0; t < TOTAL_; ++t) s += __expf(row[t] - mx);
    float inv = 1.0f / s;
    tp[0] = __expf(row[t0] - mx) * inv;
    tp[1] = __expf(row[t1] - mx) * inv;
    tp[2] = __expf(row[t2] - mx) * inv;
    tp[3] = __expf(row[t3] - mx) * inv;
    nt    = __expf(row[NT_] - mx) * inv;
    to    = __expf(row[TO_] - mx) * inv;
}

__global__ __launch_bounds__(256) void tree_dp_kernel(
    const float* __restrict__ logits, const int* __restrict__ tgt,
    float* __restrict__ loss)
{
    __shared__ float P[L_][K_];
    __shared__ float O[L_];

    const int b   = blockIdx.x;
    const int tid = threadIdx.x;
    const int t0 = tgt[0], t1 = tgt[1], t2 = tgt[2], t3 = tgt[3];

    for (int leaf = tid; leaf < L_; leaf += 256) {
        const float* row = logits + ((size_t)b * N_ + leaf) * TOTAL_;
        float tp[K_], nt, to;
        node_probs(row, t0, t1, t2, t3, tp, nt, to);
        P[leaf][0] = tp[0]; P[leaf][1] = tp[1];
        P[leaf][2] = tp[2]; P[leaf][3] = tp[3];
        O[leaf] = 1.0f - (tp[0] + tp[1] + tp[2] + tp[3] + nt + to);
    }
    __syncthreads();

    int off = L_;
    for (int m = L_ / 2; m >= 1; m >>= 1) {
        float np[K_], no = 0.0f;
        const bool act = (tid < m);
        if (act) {
            const float* row = logits + ((size_t)b * N_ + off + tid) * TOTAL_;
            float tp[K_], nt, to;
            node_probs(row, t0, t1, t2, t3, tp, nt, to);
            float ox = O[2 * tid], oy = O[2 * tid + 1];
            #pragma unroll
            for (int k = 0; k < K_; ++k) {
                float x = P[2 * tid][k], y = P[2 * tid + 1][k];
                np[k] = tp[k] + nt * (x + y - x * y);
            }
            no = 1.0f - (tp[0] + tp[1] + tp[2] + tp[3] + nt + to)
               + nt * (1.0f - (1.0f - ox) * (1.0f - oy));
        }
        __syncthreads();
        if (act) {
            P[tid][0] = np[0]; P[tid][1] = np[1];
            P[tid][2] = np[2]; P[tid][3] = np[3];
            O[tid] = no;
        }
        __syncthreads();
        off += m;
    }

    if (tid == 0) {
        float ls = 0.0f;
        #pragma unroll
        for (int k = 0; k < K_; ++k) ls -= logf(fmaxf(P[0][k], EPS_));
        ls -= logf(fmaxf(1.0f - O[0], EPS_));
        atomicAdd(loss, ls * (1.0f / (float)B_));
    }
}

__global__ void zero_loss_kernel(float* loss) { *loss = 0.0f; }

// ---------------------------------------------------------------------------
// launcher
// ---------------------------------------------------------------------------
extern "C" void kernel_launch(void* const* d_in, const int* in_sizes, int n_in,
                              void* d_out, int out_size, void* d_ws, size_t ws_size,
                              hipStream_t stream)
{
    (void)in_sizes; (void)n_in; (void)out_size;

    const float* E   = (const float*)d_in[0];
    const float* W1  = (const float*)d_in[1];
    const float* b1  = (const float*)d_in[2];
    const float* W2  = (const float*)d_in[3];
    const float* b2  = (const float*)d_in[4];
    const int*   tgt = (const int*)  d_in[5];

    float* out    = (float*)d_out;
    float* logits = out;                                // [B, N, TOTAL] flat
    float* loss   = out + (size_t)M_ * TOTAL_;          // last element

    const size_t efrag = (size_t)M_ * H_;               // elements per E array
    const size_t wfrag = (size_t)H_ * I_;               // elements per W array
    const size_t need  = (2 * efrag + 2 * wfrag) * sizeof(__bf16);  // ~105 MB

    zero_loss_kernel<<<1, 1, 0, stream>>>(loss);

    if (ws_size >= need) {
        __bf16* Ef_hi = (__bf16*)d_ws;
        __bf16* Ef_lo = Ef_hi + efrag;
        __bf16* Wf_hi = Ef_lo + efrag;
        __bf16* Wf_lo = Wf_hi + wfrag;

        prep_e_kernel<<<(int)((efrag + 255) / 256), 256, 0, stream>>>(E, Ef_hi, Ef_lo);
        prep_w_kernel<<<(int)((wfrag + 255) / 256), 256, 0, stream>>>(W1, Wf_hi, Wf_lo);
        fused_mlp_pre_kernel<<<M_ / (16 * MSUB_), 256, 0, stream>>>(Ef_hi, Ef_lo,
                                                                    Wf_hi, Wf_lo,
                                                                    b1, W2, b2, logits);
    } else {
        fused_mlp_kernel<<<M_ / 16, 256, 0, stream>>>(E, W1, b1, W2, b2, logits);
    }

    tree_dp_kernel<<<B_, 256, 0, stream>>>(logits, tgt, loss);
}
